// MultiHeadAttention_56891136803059
// MI455X (gfx1250) — matrix operand order repaired and checked by reading, not compiled, not run on hardware
//
#include <hip/hip_runtime.h>
#include <hip/hip_bf16.h>

typedef __bf16 bf16_t;
typedef __attribute__((ext_vector_type(8)))  __bf16 v8bf;
typedef __attribute__((ext_vector_type(16))) __bf16 v16bf;
typedef __attribute__((ext_vector_type(8)))  float  v8f;
typedef __attribute__((ext_vector_type(4)))  unsigned int v4u;
typedef __attribute__((ext_vector_type(8)))  int v8i;
typedef __attribute__((ext_vector_type(4)))  int v4i;

#if defined(__gfx1250__) && __has_builtin(__builtin_amdgcn_tensor_load_to_lds) && \
    __has_builtin(__builtin_amdgcn_s_wait_tensorcnt)
#define HAVE_TDM 1
#else
#define HAVE_TDM 0
#endif

static __device__ __forceinline__ v8f wmma_bf16(v16bf a, v16bf b, v8f c) {
  return __builtin_amdgcn_wmma_f32_16x16x32_bf16(false, a, false, b, (short)0, c,
                                                 false, false);
}

// Interleaved 16-bit A-matrix fragment (ISA 7.12.2):
// lane<16: K = {half*8+0..7, 16+half*8+0..7}; two aligned 16B loads combined.
static __device__ __forceinline__ v16bf combine8(v8bf lo, v8bf hi) {
  v16bf r;
#pragma unroll
  for (int i = 0; i < 8; ++i) { r[i] = lo[i]; r[8 + i] = hi[i]; }
  return r;
}

// ---------------------------------------------------------------------------
// Elementwise f32 -> bf16
// ---------------------------------------------------------------------------
__global__ __launch_bounds__(256)
void cvt_f32_bf16(const float* __restrict__ in, bf16_t* __restrict__ out, int n) {
  int i = blockIdx.x * 256 + threadIdx.x;
  if (i < n) out[i] = (bf16_t)in[i];
}

// ---------------------------------------------------------------------------
// W [K x N] f32  ->  Wt [N x K] bf16   (32x32 tiles via LDS)
// ---------------------------------------------------------------------------
__global__ __launch_bounds__(256)
void transpose_w_bf16(const float* __restrict__ W, bf16_t* __restrict__ Wt,
                      int K, int N) {
  __shared__ float tile[32][33];
  const int tx = threadIdx.x & 31;
  const int ty = threadIdx.x >> 5;
  const int k0 = blockIdx.x * 32;
  const int n0 = blockIdx.y * 32;
#pragma unroll
  for (int i = 0; i < 4; ++i)
    tile[ty + i * 8][tx] = W[(size_t)(k0 + ty + i * 8) * N + n0 + tx];
  __syncthreads();
#pragma unroll
  for (int i = 0; i < 4; ++i)
    Wt[(size_t)(n0 + ty + i * 8) * K + k0 + tx] = (bf16_t)tile[tx][ty + i * 8];
}

// ---------------------------------------------------------------------------
// TDM: DMA a 2D bf16 tile (tile_w x tile_h, row stride = stride elems) from
// global memory into LDS, inserting 8 DWORDs of pad after every 16 DWORDs
// (64B row -> LDS row stride 96B = 48 bf16, matching LDSS below).
// ---------------------------------------------------------------------------
#if HAVE_TDM
static __device__ __forceinline__ void tdm_load_2d(unsigned lds_byte_addr,
                                                   const void* gptr,
                                                   int tile_w, int tile_h,
                                                   int stride_elems) {
  const unsigned long long ga = (unsigned long long)gptr;
  const v4u g0 = {
      1u,                                   // count=1, user descriptor
      lds_byte_addr,                        // lds_addr [63:32]
      (unsigned)(ga & 0xffffffffu),         // global_addr lo
      (unsigned)((ga >> 32) & 0x01ffffffu) | (2u << 30)  // addr hi | type=2
  };
  const v8i g1 = {
      (1 << 16) |                           // data_size = 2 bytes
      (1 << 20) |                           // pad_enable
      (3 << 22) |                           // pad_interval: 16 DWORDs
      (7 << 25),                            // pad_amount: 8 DWORDs
      (int)((unsigned)tile_w << 16),        // tensor_dim0[15:0] @ [31:16]
      (int)((unsigned)tile_h << 16),        // dim0 hi=0, tensor_dim1 lo
      (int)((unsigned)tile_w << 16),        // dim1 hi=0, tile_dim0
      tile_h,                               // tile_dim1 (tile_dim2 = 0)
      stride_elems,                         // tensor_dim0_stride lo
      0, 0                                  // stride hi / dim1_stride (unused 2D)
  };
  const v4i zero4 = {0, 0, 0, 0};
  const v8i zero8 = {0, 0, 0, 0, 0, 0, 0, 0};
  // 6-arg form (this toolchain): (v4u g0, v8i g1, v4i g2, v4i g3, v8i, i32 cpol)
  __builtin_amdgcn_tensor_load_to_lds(g0, g1, zero4, zero4, zero8, 0);
}
#endif

// ---------------------------------------------------------------------------
// C = A (MxK bf16 row-major) x Bt^T (Bt is NxK bf16 row-major)
// Block tile 128x64, 8 waves (4x2), 32x32 per wave, K-step 32.
// TDM path: double-buffered async pipeline — wave 0 issues the next tile's
// DMA right after the single per-iteration barrier, so the Tensor Data Mover
// streams tile i+1 while all 8 waves run WMMAs on tile i.
// MODE 1: bf16 out, head-major [B,H,T,64]; MODE 2: bf16 out [B,H,64,T];
// MODE 3: f32 out row-major [M,N].
// ---------------------------------------------------------------------------
#define TM 128
#define TN 64
#define TK 32
#define LDSS 48   // padded LDS row stride (bf16): 96B, matches TDM padding

template <int MODE>
__global__ __launch_bounds__(256)
void gemm_bf16_wmma(const bf16_t* __restrict__ A,
                    const bf16_t* __restrict__ Bt,
                    void* __restrict__ Cout,
                    int M, int N, int K) {
#if HAVE_TDM
  __shared__ alignas(32) bf16_t lA[2][TM * LDSS];
  __shared__ alignas(32) bf16_t lB[2][TN * LDSS];
#else
  __shared__ alignas(32) bf16_t lA[1][TM * LDSS];
  __shared__ alignas(32) bf16_t lB[1][TN * LDSS];
#endif

  const int tid  = threadIdx.x;
  const int lane = tid & 31;
  const int wave = tid >> 5;
  const int wm   = wave & 3;
  const int wn   = wave >> 2;
  const int half = lane >> 4;
  const int l15  = lane & 15;

  const int bm = blockIdx.x * TM;
  const int bn = blockIdx.y * TN;

  v8f acc[2][2] = {};

#if HAVE_TDM
  if (wave == 0) {   // prologue: start DMA of tile 0 into buffer 0
    tdm_load_2d((unsigned)(size_t)&lA[0][0], A + (size_t)bm * K, TK, TM, K);
    tdm_load_2d((unsigned)(size_t)&lB[0][0], Bt + (size_t)bn * K, TK, TN, K);
  }
#endif

  for (int k0 = 0; k0 < K; k0 += TK) {
#if HAVE_TDM
    const int cur = (k0 / TK) & 1;
    if (wave == 0) __builtin_amdgcn_s_wait_tensorcnt(0);  // tile k0 landed
    __syncthreads();   // single barrier/iter: also separates buffer reuse
    if (wave == 0 && k0 + TK < K) {   // kick off tile k0+TK into other buffer
      tdm_load_2d((unsigned)(size_t)&lA[cur ^ 1][0],
                  A + (size_t)bm * K + k0 + TK, TK, TM, K);
      tdm_load_2d((unsigned)(size_t)&lB[cur ^ 1][0],
                  Bt + (size_t)bn * K + k0 + TK, TK, TN, K);
    }
#else
    const int cur = 0;
#pragma unroll
    for (int i = 0; i < 2; ++i) {
      const int chunk = tid * 2 + i;
      const int r = chunk >> 2;
      const int c = (chunk & 3) * 8;
      *(v8bf*)(lA[0] + r * LDSS + c) =
          *(const v8bf*)(A + (size_t)(bm + r) * K + k0 + c);
    }
    {
      const int r = tid >> 2;
      const int c = (tid & 3) * 8;
      *(v8bf*)(lB[0] + r * LDSS + c) =
          *(const v8bf*)(Bt + (size_t)(bn + r) * K + k0 + c);
    }
    if (k0 + TK < K) {
      __builtin_prefetch(A + (size_t)(bm + (tid >> 1)) * K + k0 + TK, 0, 3);
      __builtin_prefetch(Bt + (size_t)(bn + (tid >> 2)) * K + k0 + TK, 0, 3);
    }
    __syncthreads();
#endif

    v16bf af[2], bf[2];
#pragma unroll
    for (int ti = 0; ti < 2; ++ti) {
      const bf16_t* p = lA[cur] + (wm * 32 + ti * 16 + l15) * LDSS + half * 8;
      af[ti] = combine8(*(const v8bf*)p, *(const v8bf*)(p + 16));
    }
#pragma unroll
    for (int tj = 0; tj < 2; ++tj) {   // B frag: lanes 0-15 K={0..7,16..23}, 16-31 K={8..15,24..31}
      const bf16_t* q = lB[cur] + (wn * 32 + tj * 16 + l15) * LDSS + half * 8;
      bf[tj] = combine8(*(const v8bf*)q, *(const v8bf*)(q + 16));
    }
#pragma unroll
    for (int ti = 0; ti < 2; ++ti)
#pragma unroll
      for (int tj = 0; tj < 2; ++tj)
        acc[ti][tj] = wmma_bf16(af[ti], bf[tj], acc[ti][tj]);
#if !HAVE_TDM
    __syncthreads();
#endif
  }

  // C layout: VGPR r -> row r + 8*half, lane&15 -> col.
#pragma unroll
  for (int ti = 0; ti < 2; ++ti) {
#pragma unroll
    for (int tj = 0; tj < 2; ++tj) {
      const int row0 = bm + wm * 32 + ti * 16 + half * 8;
      const int col  = bn + wn * 32 + tj * 16 + l15;
#pragma unroll
      for (int r = 0; r < 8; ++r) {
        const int row = row0 + r;
        const float v = acc[ti][tj][r];
        if (MODE == 3) {
          ((float*)Cout)[(size_t)row * N + col] = v;
        } else {
          const int b = row >> 11, t = row & 2047;   // T = 2048
          const int h = col >> 6,  hd = col & 63;    // H = 16, hd = 64
          size_t idx;
          if (MODE == 1) idx = ((size_t)(b * 16 + h) * 2048 + t) * 64 + hd;
          else           idx = ((size_t)(b * 16 + h) * 64 + hd) * 2048 + t;
          ((bf16_t*)Cout)[idx] = (bf16_t)v;
        }
      }
    }
  }
}

// ---------------------------------------------------------------------------
// Causal flash attention, S computed TRANSPOSED (S^T = K Q^T) so that:
//  - softmax stats are per-lane scalars (keys live in registers),
//  - P^T's C-layout IS the A-fragment of P for O += P*V (no LDS, no shuffles).
// 1 block = (b,h, 64 q rows), 4 waves x 16 rows.  LDS-free.
// Qh,Kh: [B,H,T,64] bf16;  Vt: [B,H,64,T] bf16;  ctx: [B*T,1024] bf16.
// ---------------------------------------------------------------------------
__global__ __launch_bounds__(128)
void flash_attn(const bf16_t* __restrict__ Qh, const bf16_t* __restrict__ Kh,
                const bf16_t* __restrict__ Vt, bf16_t* __restrict__ ctx) {
  constexpr int T = 2048;
  const int tid  = threadIdx.x;
  const int lane = tid & 31;
  const int wave = tid >> 5;
  const int half = lane >> 4;
  const int l15  = lane & 15;

  const int blk = blockIdx.x;
  const int qt  = blk & 31;          // T/64 = 32
  const int bh  = blk >> 5;
  const int b   = bh >> 4;           // H = 16
  const int h   = bh & 15;

  const int q0 = qt * 64 + wave * 16;
  const size_t hoff = (size_t)bh * T * 64;
  const bf16_t* Qp = Qh + hoff;
  const bf16_t* Kp = Kh + hoff;
  const bf16_t* Vp = Vt + hoff;

  // Q^T as B fragments: lane = query, elements = hd (interleaved, two 16B loads)
  v16bf bq[2];
#pragma unroll
  for (int c = 0; c < 2; ++c)
    bq[c] = combine8(*(const v8bf*)(Qp + (size_t)(q0 + l15) * 64 + c * 32 + half * 8),
                     *(const v8bf*)(Qp + (size_t)(q0 + l15) * 64 + c * 32 + half * 8 + 16));

  v8f o[4] = {};
  float m = -1e30f, l = 0.f;
  const float scale = 0.125f;        // 1/sqrt(64)

  const int kmax = q0 + 15;
  for (int k0 = 0; k0 <= kmax; k0 += 32) {
    // S^T (32 keys x 16 queries): A = K tile rows, B = Q^T
    v8f s[2] = {};
#pragma unroll
    for (int ns = 0; ns < 2; ++ns) {
      const bf16_t* kp = Kp + (size_t)(k0 + ns * 16 + l15) * 64;
#pragma unroll
      for (int c = 0; c < 2; ++c) {
        v16bf ak = combine8(*(const v8bf*)(kp + c * 32 + half * 8),
                            *(const v8bf*)(kp + c * 32 + half * 8 + 16));
        s[ns] = wmma_bf16(ak, bq[c], s[ns]);
      }
    }
    // scale + causal mask: key = k0+ns*16+r+8*half (VGPR axis), query = q0+l15
    const int q = q0 + l15;
#pragma unroll
    for (int ns = 0; ns < 2; ++ns) {
      const int kb = k0 + ns * 16 + half * 8;
#pragma unroll
      for (int r = 0; r < 8; ++r) {
        const float v = s[ns][r] * scale;
        s[ns][r] = (kb + r <= q) ? v : -1e30f;
      }
    }
    // online softmax: in-register trees + one xor-16 pair-combine
    float pm = -1e30f;
#pragma unroll
    for (int ns = 0; ns < 2; ++ns)
#pragma unroll
      for (int r = 0; r < 8; ++r) pm = fmaxf(pm, s[ns][r]);
    pm = fmaxf(pm, __shfl_xor(pm, 16, 32));
    const float mnew = fmaxf(m, pm);
    const float corr = __expf(m - mnew);
    m = mnew;
    float ps = 0.f;
#pragma unroll
    for (int ns = 0; ns < 2; ++ns)
#pragma unroll
      for (int r = 0; r < 8; ++r) {
        s[ns][r] = __expf(s[ns][r] - mnew);
        ps += s[ns][r];
      }
    ps += __shfl_xor(ps, 16, 32);
    l = l * corr + ps;

    // P A-fragment directly from registers (no cross-lane movement needed)
    v16bf ap;
#pragma unroll
    for (int i = 0; i < 8; ++i) {
      ap[i]     = (bf16_t)s[0][i];
      ap[8 + i] = (bf16_t)s[1][i];
    }
    // broadcast per-query corr onto O's row axis: row r -> lane r + 8*half
    float corrb[8];
#pragma unroll
    for (int r = 0; r < 8; ++r) corrb[r] = __shfl(corr, r + 8 * half, 32);

#pragma unroll
    for (int j = 0; j < 4; ++j) {
      v16bf bv = combine8(*(const v8bf*)(Vp + (size_t)(j * 16 + l15) * T + k0 + half * 8),
                          *(const v8bf*)(Vp + (size_t)(j * 16 + l15) * T + k0 + half * 8 + 16));
#pragma unroll
      for (int r = 0; r < 8; ++r) o[j][r] *= corrb[r];
      o[j] = wmma_bf16(ap, bv, o[j]);
    }
  }

  // normalize rows and scatter to ctx[(b*T + t), h*64 + hd]
  float linv[8];
#pragma unroll
  for (int r = 0; r < 8; ++r) linv[r] = 1.0f / __shfl(l, r + 8 * half, 32);
#pragma unroll
  for (int j = 0; j < 4; ++j) {
    const int hd = j * 16 + l15;
#pragma unroll
    for (int r = 0; r < 8; ++r) {
      const int t = q0 + r + half * 8;
      ctx[(size_t)(b * T + t) * 1024 + h * 64 + hd] = (bf16_t)(o[j][r] * linv[r]);
    }
  }
}

// ---------------------------------------------------------------------------
extern "C" void kernel_launch(void* const* d_in, const int* in_sizes, int n_in,
                              void* d_out, int out_size, void* d_ws, size_t ws_size,
                              hipStream_t stream) {
  (void)in_sizes; (void)n_in; (void)out_size; (void)ws_size;
  const float* x  = (const float*)d_in[0];
  const float* Wq = (const float*)d_in[1];
  const float* Wk = (const float*)d_in[2];
  const float* Wv = (const float*)d_in[3];
  const float* Wo = (const float*)d_in[4];
  float* out = (float*)d_out;

  constexpr int B = 2, T = 2048, D = 1024, H = 16;
  constexpr int M = B * T;

  char* ws = (char*)d_ws;
  size_t off = 0;
  auto alloc = [&](size_t bytes) -> void* {
    void* p = ws + off;
    off += (bytes + 255) & ~(size_t)255;
    return p;
  };
  bf16_t* xb  = (bf16_t*)alloc((size_t)M * D * sizeof(bf16_t));
  bf16_t* wqt = (bf16_t*)alloc((size_t)D * D * sizeof(bf16_t));
  bf16_t* wkt = (bf16_t*)alloc((size_t)D * D * sizeof(bf16_t));
  bf16_t* wvt = (bf16_t*)alloc((size_t)D * D * sizeof(bf16_t));
  bf16_t* wot = (bf16_t*)alloc((size_t)D * D * sizeof(bf16_t));
  bf16_t* Qh  = (bf16_t*)alloc((size_t)M * D * sizeof(bf16_t));
  bf16_t* Kh  = (bf16_t*)alloc((size_t)M * D * sizeof(bf16_t));
  bf16_t* Vt  = (bf16_t*)alloc((size_t)M * D * sizeof(bf16_t));
  bf16_t* ctx = (bf16_t*)alloc((size_t)M * D * sizeof(bf16_t));

  {
    const int n = M * D;
    cvt_f32_bf16<<<dim3((n + 255) / 256), dim3(256), 0, stream>>>(x, xb, n);
  }
  {
    dim3 g(D / 32, D / 32), blk(256);
    transpose_w_bf16<<<g, blk, 0, stream>>>(Wq, wqt, D, D);
    transpose_w_bf16<<<g, blk, 0, stream>>>(Wk, wkt, D, D);
    transpose_w_bf16<<<g, blk, 0, stream>>>(Wv, wvt, D, D);
    transpose_w_bf16<<<g, blk, 0, stream>>>(Wo, wot, D, D);
  }
  dim3 gg(M / TM, D / TN), bb(256);
  gemm_bf16_wmma<1><<<gg, bb, 0, stream>>>(xb, wqt, Qh, M, D, D);
  gemm_bf16_wmma<1><<<gg, bb, 0, stream>>>(xb, wkt, Kh, M, D, D);
  gemm_bf16_wmma<2><<<gg, bb, 0, stream>>>(xb, wvt, Vt, M, D, D);
  flash_attn<<<dim3(B * H * (T / 64)), dim3(128), 0, stream>>>(Qh, Kh, Vt, ctx);
  gemm_bf16_wmma<3><<<gg, bb, 0, stream>>>(ctx, wot, out, M, D, D);
}
